// GPT_12197707121223
// MI455X (gfx1250) — matrix operand order repaired and checked by reading, not compiled, not run on hardware
//
#include <hip/hip_runtime.h>
#include <hip/hip_bf16.h>
#include <math.h>

// ---------------------------------------------------------------------------
// GPT-2 small forward pass for MI455X (gfx1250), wave32 + WMMA + TDM.
// All matmuls run through a templated, double-buffered WMMA GEMM using
// v_wmma_f32_16x16x32_f16 (f16 inputs, f32 accumulate). Weights are converted
// once per launch to f16 in B^T ([N][K]) layout so both operand fragments are
// contiguous 16B LDS reads. LN / softmax / GELU / residuals / logits are fp32.
//
// Tile staging uses the CDNA5 Tensor Data Mover: wave 0 issues one
// tensor_load_to_lds descriptor per operand tile (A: 128x32 f16, B: BNx32 f16)
// for the next buffer while all 8 waves run WMMAs on the current buffer.
// The D# pad feature (pad_interval=16 DWORDs, pad_amount=4 DWORDs) reproduces
// the 40-f16 LDS row stride (bank-conflict padding) in hardware.
// Synchronization: wave0 s_wait_tensorcnt(0) + workgroup barrier per K-step.
// The LM head GEMM (N=50257, needs N-bound guards) keeps a register-staged
// global_load_b128 / ds_store_b128 path.
// ---------------------------------------------------------------------------

typedef __attribute__((ext_vector_type(16))) _Float16 v16h;
typedef __attribute__((ext_vector_type(8)))  float    v8f;
typedef __attribute__((ext_vector_type(4)))  unsigned int u32x4;
typedef __attribute__((ext_vector_type(4)))  int      i32x4;
typedef __attribute__((ext_vector_type(8)))  int      i32x8;

#define TB   2
#define TS   1024
#define TV   50257
#define TD   768
#define TH   12
#define TL   12
#define TDN  64
#define TDFF 3072
#define TNQ  768     // H * DN

#define BM   128
#define KT   32
#define LDSP 40      // LDS row stride in f16 (32 + 8 pad, keeps 16B alignment)

enum { EPI_SCORES = 0, EPI_GELU = 2, EPI_RESID = 3, EPI_QKV = 4, EPI_PV = 5, EPI_F32 = 6 };

struct GemmP {
  const _Float16* A;  long long sAz; int lda;   // A  [M][K], batched by z
  const _Float16* BT; long long sBz; int ldb;   // B^T[N][K], batched by z
  const float* bias;                            // per-output-column (or packed qkv)
  const float* resid; int ldr;                  // fp32 residual for EPI_RESID
  float*    out32; long long sCz; int ldc;
  _Float16* out16;
  int M, N, K;
  float scale;
  int epi;
};

// Issue one TDM 2D tile load: KT x rows f16 tile, row stride = stride_elems,
// destination LDS rows padded to LDSP f16 (pad 4 DWORDs every 16 DWORDs).
// D# layout per CDNA5 ISA (group0 128b, group1 256b).
__device__ __forceinline__
void tdm_load_tile(unsigned lds_off, const _Float16* gptr, int rows, int stride_elems) {
  unsigned long long ga = (unsigned long long)(size_t)gptr;
  u32x4 g0;
  g0[0] = 1u;                                   // count=1, user descriptor
  g0[1] = lds_off;                              // lds_addr (bytes)
  g0[2] = (unsigned)ga;                         // global_addr[31:0]
  g0[3] = (unsigned)((ga >> 32) & 0x01FFFFFFu)  // global_addr[56:32]
          | 0x80000000u;                        // type=2 ("image")
  i32x8 g1;
  g1[0] = (int)((1u << 16)                      // data_size = 2 bytes
                | (1u << 20)                    // pad_enable
                | (3u << 22)                    // pad_interval: 16 DWORDs
                | (3u << 25));                  // pad_amount:    4 DWORDs
  g1[1] = (int)((unsigned)KT << 16);            // tensor_dim0[15:0] (atomic_addr=0)
  g1[2] = (int)((unsigned)rows << 16);          // tensor_dim0[31:16]=0 | tensor_dim1[15:0]
  g1[3] = (int)((unsigned)KT << 16);            // tensor_dim1[31:16]=0 | tile_dim0=KT
  g1[4] = (int)((unsigned)rows & 0xFFFFu);      // tile_dim1=rows, tile_dim2=0
  g1[5] = (int)(unsigned)stride_elems;          // tensor_dim0_stride[31:0]
  g1[6] = 0;                                    // stride0[47:32]=0, stride1[15:0]=0
  g1[7] = 0;
  i32x4 z4 = {0, 0, 0, 0};
#if defined(__clang_major__) && (__clang_major__ >= 23)
  i32x8 z8 = {0, 0, 0, 0, 0, 0, 0, 0};
  __builtin_amdgcn_tensor_load_to_lds(g0, g1, z4, z4, z8, 0);
#else
  __builtin_amdgcn_tensor_load_to_lds(g0, g1, z4, z4, 0);
#endif
}

// BN_T: block tile N (128 or 64). Wave tile is 32 x (BN_T/2).
// GN:   guard N bounds (LM head only).  TDMEN: use Tensor Data Mover staging.
template <int EPI, int BN_T, bool GN, bool TDMEN>
__global__ __launch_bounds__(256)
void wmma_gemm(GemmP P) {
  static_assert(!(TDMEN && GN), "TDM path has no N-bound guard");
  constexpr int WN = BN_T / 2;     // per-wave N extent (2 waves along N)
  constexpr int NT = WN / 16;      // 16x16 tiles along N per wave
  __shared__ _Float16 As[2 * BM * LDSP];
  __shared__ _Float16 Bs[2 * BN_T * LDSP];

  const int tid  = threadIdx.x;
  const int wave = tid >> 5;
  const int lane = tid & 31;
  const int hl   = lane >> 4;      // lane half (WMMA layout)
  const int lrow = lane & 15;
  const int waveM = (wave & 3) * 32;   // 4 waves along M
  const int waveN = (wave >> 2) * WN;  // 2 waves along N
  const int blockM = blockIdx.y * BM;
  const int blockN = blockIdx.x * BN_T;
  const int z = blockIdx.z;

  const _Float16* Ab  = P.A  + (size_t)z * (size_t)P.sAz;
  const _Float16* BTb = P.BT + (size_t)z * (size_t)P.sBz;

  v8f zv = {};
  v8f acc[2][NT];
#pragma unroll
  for (int i = 0; i < 2; ++i)
#pragma unroll
    for (int j = 0; j < NT; ++j) acc[i][j] = zv;

  // one K-step of WMMAs from LDS buffer `buf`
  auto mma_step = [&](int buf) {
    const _Float16* Ash = &As[(size_t)buf * (BM * LDSP)];
    const _Float16* Bsh = &Bs[(size_t)buf * (BN_T * LDSP)];
    v16h af[2], bf[NT];
#pragma unroll
    for (int mt = 0; mt < 2; ++mt) {
      // A lane: row M = tile+lrow; K run [hl*16, +16)
      const _Float16* p = &Ash[(waveM + mt * 16 + lrow) * LDSP + hl * 16];
      ((uint4*)&af[mt])[0] = *(const uint4*)p;
      ((uint4*)&af[mt])[1] = *(const uint4*)(p + 8);
    }
#pragma unroll
    for (int nt = 0; nt < NT; ++nt) {
      // B lane: col N = tile+lrow; K run [hl*16, +16)
      const _Float16* p = &Bsh[(waveN + nt * 16 + lrow) * LDSP + hl * 16];
      ((uint4*)&bf[nt])[0] = ((const uint4*)p)[0];
      ((uint4*)&bf[nt])[1] = ((const uint4*)p)[1];
    }
#pragma unroll
    for (int mt = 0; mt < 2; ++mt)
#pragma unroll
      for (int nt = 0; nt < NT; ++nt)
        acc[mt][nt] = __builtin_amdgcn_wmma_f32_16x16x32_f16(
            false, af[mt], false, bf[nt], (short)0, acc[mt][nt], false, false);
  };

  if constexpr (TDMEN) {
    // ---- Tensor Data Mover staging, double buffered, 1 barrier / K-step ----
    const unsigned asb = (unsigned)(size_t)(void*)As;
    const unsigned bsb = (unsigned)(size_t)(void*)Bs;
    const _Float16* aTile = Ab + (size_t)blockM * P.lda;
    const _Float16* bTile = BTb + (size_t)blockN * P.ldb;
    if (wave == 0) {
      tdm_load_tile(asb, aTile, BM, P.lda);
      tdm_load_tile(bsb, bTile, BN_T, P.ldb);
    }
    int buf = 0;
    for (int k0 = 0; k0 < P.K; k0 += KT) {
      if (wave == 0) __builtin_amdgcn_s_wait_tensorcnt(0);
      __syncthreads();  // publish tile(buf); also fences reads of buf^1 (iter-1)
      if (wave == 0 && (k0 + KT < P.K)) {
        int nb = buf ^ 1;
        tdm_load_tile(asb + (unsigned)nb * (BM * LDSP * 2),
                      aTile + (k0 + KT), BM, P.lda);
        tdm_load_tile(bsb + (unsigned)nb * (BN_T * LDSP * 2),
                      bTile + (k0 + KT), BN_T, P.ldb);
      }
      mma_step(buf);
      buf ^= 1;
    }
  } else {
    // ---- register-staged path (used by the N-guarded LM head GEMM) ----
    const int rowA = tid >> 1;
    const int colA = (tid & 1) << 4;
    const _Float16* aptr = Ab + (size_t)(blockM + rowA) * P.lda + colA;

    int rowB, colB;
    if constexpr (BN_T == 128) { rowB = tid >> 1; colB = (tid & 1) << 4; }
    else                       { rowB = tid >> 2; colB = (tid & 3) << 3; }
    const int gnB = blockN + rowB;
    const _Float16* bptr = BTb + (size_t)gnB * P.ldb + colB;
    const bool bok = (!GN) || (gnB < P.N);

    uint4 ra0, ra1;
    uint4 rb0 = make_uint4(0, 0, 0, 0), rb1 = make_uint4(0, 0, 0, 0);

    auto fetch = [&](int k0) {
      const uint4* ap = (const uint4*)(aptr + k0);
      ra0 = ap[0]; ra1 = ap[1];
      if (bok) {
        const uint4* bp = (const uint4*)(bptr + k0);
        rb0 = bp[0];
        if constexpr (BN_T == 128) rb1 = bp[1];
      } else {
        rb0 = make_uint4(0, 0, 0, 0);
        if constexpr (BN_T == 128) rb1 = make_uint4(0, 0, 0, 0);
      }
    };
    auto stage = [&](int buf) {
      uint4* da = (uint4*)&As[(size_t)buf * (BM * LDSP) + rowA * LDSP + colA];
      da[0] = ra0; da[1] = ra1;
      uint4* db = (uint4*)&Bs[(size_t)buf * (BN_T * LDSP) + rowB * LDSP + colB];
      db[0] = rb0;
      if constexpr (BN_T == 128) db[1] = rb1;
    };

    fetch(0);
    int buf = 0;
    for (int k0 = 0; k0 < P.K; k0 += KT) {
      stage(buf);
      __syncthreads();
      if (k0 + KT < P.K) fetch(k0 + KT);          // overlaps with WMMAs below
      if (k0 + 2 * KT < P.K) {
        __builtin_prefetch(aptr + k0 + 2 * KT, 0, 1);
        if (bok) __builtin_prefetch(bptr + k0 + 2 * KT, 0, 1);
      }
      mma_step(buf);
      buf ^= 1;
    }
  }

  // ---- epilogue: C lane layout: col = lrow, rows = hl*8 + r ----
#pragma unroll
  for (int mt = 0; mt < 2; ++mt) {
#pragma unroll
    for (int nt = 0; nt < NT; ++nt) {
      const int gn = blockN + waveN + nt * 16 + lrow;
      if (GN && gn >= P.N) continue;
#pragma unroll
      for (int r = 0; r < 8; ++r) {
        const int gm = blockM + waveM + mt * 16 + hl * 8 + r;
        float v = acc[mt][nt][r];
        if constexpr (EPI == EPI_SCORES) {
          P.out32[(size_t)z * (size_t)P.sCz + (size_t)gm * P.ldc + gn] = v * P.scale;
        } else if constexpr (EPI == EPI_GELU) {
          float t = v + P.bias[gn];
          P.out16[(size_t)gm * P.ldc + gn] =
              (_Float16)(0.5f * t * (1.0f + erff(t * 0.70710678118654752f)));
        } else if constexpr (EPI == EPI_RESID) {
          float t = v + P.bias[gn] + P.resid[(size_t)gm * P.ldr + gn];
          P.out32[(size_t)gm * P.ldc + gn] = t;
        } else if constexpr (EPI == EPI_QKV) {
          float t = v + P.bias[z * P.N + gn];
          int bb = gm / TS, ss = gm % TS;
          int hh = gn / TDN, nn = gn % TDN;
          _Float16* region = P.out16 + (size_t)z * ((size_t)P.M * (size_t)P.N);
          if (z < 2)  // q,k : [B,H,S,DN]
            region[(((size_t)bb * TH + hh) * TS + ss) * TDN + nn] = (_Float16)t;
          else        // v   : [B,H,DN,S]  (pre-transposed for the PV GEMM)
            region[(((size_t)bb * TH + hh) * TDN + nn) * TS + ss] = (_Float16)t;
        } else if constexpr (EPI == EPI_PV) {
          int bb = z / TH, hh = z % TH;
          P.out16[((size_t)bb * TS + gm) * TD + hh * TDN + gn] = (_Float16)v;
        } else {  // EPI_F32 (LM head)
          P.out32[(size_t)gm * (size_t)P.ldc + gn] = v;
        }
      }
    }
  }
}

// ---------------------------------------------------------------------------
// fp32 helper kernels
// ---------------------------------------------------------------------------

__global__ void embed_kernel(const int* x, const float* wte, const float* wpe, float* h) {
  size_t idx = (size_t)blockIdx.x * 256 + threadIdx.x;
  if (idx >= (size_t)TB * TS * TD) return;
  size_t t = idx / TD;
  int d = (int)(idx % TD);
  int s = (int)(t % TS);
  h[idx] = wte[(size_t)x[t] * TD + d] + wpe[(size_t)s * TD + d];
}

__global__ __launch_bounds__(256)
void ln_kernel(const float* h, const float* g, const float* b, _Float16* out, int D) {
  __shared__ float r1[256], r2[256];
  const int row = blockIdx.x;
  const float* xr = h + (size_t)row * D;
  float s = 0.f, ss = 0.f;
  for (int d = threadIdx.x; d < D; d += 256) { float v = xr[d]; s += v; ss += v * v; }
  r1[threadIdx.x] = s; r2[threadIdx.x] = ss;
  __syncthreads();
  for (int st = 128; st > 0; st >>= 1) {
    if (threadIdx.x < st) {
      r1[threadIdx.x] += r1[threadIdx.x + st];
      r2[threadIdx.x] += r2[threadIdx.x + st];
    }
    __syncthreads();
  }
  float mean = r1[0] / D;
  float var  = r2[0] / D - mean * mean;
  float rstd = rsqrtf(var + 1e-5f);
  _Float16* orow = out + (size_t)row * D;
  for (int d = threadIdx.x; d < D; d += 256)
    orow[d] = (_Float16)((xr[d] - mean) * rstd * g[d] + b[d]);
}

__global__ __launch_bounds__(256)
void softmax_kernel(const float* sc, _Float16* p, const int* x) {
  __shared__ float red[256];
  const int q  = blockIdx.x;
  const int bh = blockIdx.y;
  const int b  = bh / TH;
  const float* row = sc + ((size_t)bh * TS + q) * TS;
  _Float16*   prow = p  + ((size_t)bh * TS + q) * TS;
  float v[4];
  float mx = -3.4e38f;
#pragma unroll
  for (int i = 0; i < 4; ++i) {
    int k = i * 256 + threadIdx.x;
    float t = row[k];
    if (x[b * TS + k] == 0 || k > q) t += -1.0e9f;  // pad-key OR strict causal
    v[i] = t;
    mx = fmaxf(mx, t);
  }
  red[threadIdx.x] = mx; __syncthreads();
  for (int st = 128; st > 0; st >>= 1) {
    if (threadIdx.x < st) red[threadIdx.x] = fmaxf(red[threadIdx.x], red[threadIdx.x + st]);
    __syncthreads();
  }
  mx = red[0]; __syncthreads();
  float s = 0.f;
#pragma unroll
  for (int i = 0; i < 4; ++i) { v[i] = expf(v[i] - mx); s += v[i]; }
  red[threadIdx.x] = s; __syncthreads();
  for (int st = 128; st > 0; st >>= 1) {
    if (threadIdx.x < st) red[threadIdx.x] += red[threadIdx.x + st];
    __syncthreads();
  }
  float inv = 1.0f / red[0];
#pragma unroll
  for (int i = 0; i < 4; ++i)
    prow[i * 256 + threadIdx.x] = (_Float16)(v[i] * inv);
}

// transpose + fp32->f16: in [L][R][C] -> out [L][C][R]  (gives B^T layout)
__global__ void convT_kernel(const float* in, _Float16* out, int R, int C, size_t total) {
  size_t idx = (size_t)blockIdx.x * 256 + threadIdx.x;
  if (idx >= total) return;
  size_t rc = (size_t)R * C;
  size_t l = idx / rc, rem = idx % rc;
  int r = (int)(rem / C), c = (int)(rem % C);
  out[l * rc + (size_t)c * R + r] = (_Float16)in[idx];
}

// wq/wk/wv [L][H][D][DN] -> wqkvT [L][3][H*DN][D]  (packed B^T for the QKV GEMM)
__global__ void conv_qkv_kernel(const float* w, _Float16* out, int z, size_t total) {
  size_t idx = (size_t)blockIdx.x * 256 + threadIdx.x;
  if (idx >= total) return;
  size_t hdn = (size_t)TH * TD * TDN;
  size_t l = idx / hdn, rem = idx % hdn;
  int h = (int)(rem / ((size_t)TD * TDN));
  size_t rem2 = rem % ((size_t)TD * TDN);
  int d = (int)(rem2 / TDN), n = (int)(rem2 % TDN);
  out[(((size_t)l * 3 + z) * TNQ + (size_t)h * TDN + n) * TD + d] = (_Float16)w[idx];
}

__global__ void pack_bias_kernel(const float* bq, const float* bk, const float* bv,
                                 float* out, size_t total) {
  size_t idx = (size_t)blockIdx.x * 256 + threadIdx.x;
  if (idx >= total) return;
  size_t per = (size_t)3 * TNQ;
  size_t l = idx / per, rem = idx % per;
  int z = (int)(rem / TNQ), j = (int)(rem % TNQ);
  const float* s = (z == 0) ? bq : (z == 1) ? bk : bv;
  out[idx] = s[l * TNQ + j];
}

// ---------------------------------------------------------------------------
// launch
// ---------------------------------------------------------------------------

extern "C" void kernel_launch(void* const* d_in, const int* in_sizes, int n_in,
                              void* d_out, int out_size, void* d_ws, size_t ws_size,
                              hipStream_t stream) {
  (void)in_sizes; (void)n_in; (void)out_size; (void)ws_size;

  const int*   x     = (const int*)  d_in[0];
  const float* wte   = (const float*)d_in[1];
  const float* wpe   = (const float*)d_in[2];
  const float* wq    = (const float*)d_in[3];
  const float* bq    = (const float*)d_in[4];
  const float* wk    = (const float*)d_in[5];
  const float* bk    = (const float*)d_in[6];
  const float* wv    = (const float*)d_in[7];
  const float* bv    = (const float*)d_in[8];
  const float* wo    = (const float*)d_in[9];
  const float* bo    = (const float*)d_in[10];
  const float* ln1g  = (const float*)d_in[11];
  const float* ln1b  = (const float*)d_in[12];
  const float* ln2g  = (const float*)d_in[13];
  const float* ln2b  = (const float*)d_in[14];
  const float* fc1w  = (const float*)d_in[15];
  const float* fc1b  = (const float*)d_in[16];
  const float* fc2w  = (const float*)d_in[17];
  const float* fc2b  = (const float*)d_in[18];
  const float* lnfg  = (const float*)d_in[19];
  const float* lnfb  = (const float*)d_in[20];
  const float* headw = (const float*)d_in[21];
  float* out = (float*)d_out;

  // ---- workspace carving (~435 MB total) ----
  char* ws = (char*)d_ws;
  size_t off = 0;
  auto carve = [&](size_t bytes) -> void* {
    void* p = ws + off;
    off = (off + bytes + 255) & ~(size_t)255;
    return p;
  };
  _Float16* wqkvT = (_Float16*)carve((size_t)TL * 3 * TNQ * TD * 2);
  float*    bqkv  = (float*)   carve((size_t)TL * 3 * TNQ * 4);
  _Float16* woT   = (_Float16*)carve((size_t)TL * TD * TD * 2);
  _Float16* fc1T  = (_Float16*)carve((size_t)TL * TDFF * TD * 2);
  _Float16* fc2T  = (_Float16*)carve((size_t)TL * TD * TDFF * 2);
  _Float16* headT = (_Float16*)carve((size_t)TV * TD * 2);
  float*    hbuf  = (float*)   carve((size_t)TB * TS * TD * 4);
  _Float16* a16   = (_Float16*)carve((size_t)TB * TS * TD * 2);
  _Float16* qkv16 = (_Float16*)carve((size_t)3 * TB * TH * TS * TDN * 2);
  float*    sc32  = (float*)   carve((size_t)TB * TH * TS * TS * 4);
  _Float16* p16   = (_Float16*)carve((size_t)TB * TH * TS * TS * 2);
  _Float16* o16   = (_Float16*)carve((size_t)TB * TS * TD * 2);
  _Float16* ff16  = (_Float16*)carve((size_t)TB * TS * TDFF * 2);

  auto cdiv = [](size_t a, size_t b) { return (unsigned)((a + b - 1) / b); };

  // ---- one-time per-launch weight conversion to f16 B^T layout ----
  {
    size_t tot = (size_t)TL * TD * TD;
    convT_kernel<<<cdiv(tot, 256), 256, 0, stream>>>(wo, woT, TD, TD, tot);
    tot = (size_t)TL * TD * TDFF;
    convT_kernel<<<cdiv(tot, 256), 256, 0, stream>>>(fc1w, fc1T, TD, TDFF, tot);
    tot = (size_t)TL * TDFF * TD;
    convT_kernel<<<cdiv(tot, 256), 256, 0, stream>>>(fc2w, fc2T, TDFF, TD, tot);
    tot = (size_t)TV * TD;
    convT_kernel<<<cdiv(tot, 256), 256, 0, stream>>>(headw, headT, TD, TV, tot);
    tot = (size_t)TL * TH * TD * TDN;
    conv_qkv_kernel<<<cdiv(tot, 256), 256, 0, stream>>>(wq, wqkvT, 0, tot);
    conv_qkv_kernel<<<cdiv(tot, 256), 256, 0, stream>>>(wk, wqkvT, 1, tot);
    conv_qkv_kernel<<<cdiv(tot, 256), 256, 0, stream>>>(wv, wqkvT, 2, tot);
    tot = (size_t)TL * 3 * TNQ;
    pack_bias_kernel<<<cdiv(tot, 256), 256, 0, stream>>>(bq, bk, bv, bqkv, tot);
  }

  // ---- embeddings ----
  embed_kernel<<<cdiv((size_t)TB * TS * TD, 256), 256, 0, stream>>>(x, wte, wpe, hbuf);

  const float scale = 0.125f;  // 1/sqrt(DN)
  const long long bhStride = (long long)TS * TDN;

  for (int l = 0; l < TL; ++l) {
    // LN1 -> a16 (f16)
    ln_kernel<<<TB * TS, 256, 0, stream>>>(hbuf, ln1g + (size_t)l * TD,
                                           ln1b + (size_t)l * TD, a16, TD);
    // QKV projection: z in {0,1,2} selects packed weight/bias slab
    {
      GemmP P{};
      P.A = a16; P.sAz = 0; P.lda = TD;
      P.BT = wqkvT + (size_t)l * 3 * TNQ * TD; P.sBz = (long long)TNQ * TD; P.ldb = TD;
      P.bias = bqkv + (size_t)l * 3 * TNQ;
      P.out16 = qkv16;
      P.M = TB * TS; P.N = TNQ; P.K = TD;
      dim3 g(TNQ / 128, (TB * TS) / BM, 3);
      wmma_gemm<EPI_QKV, 128, false, true><<<g, 256, 0, stream>>>(P);
    }
    // scores = q . k^T * scale, batched over (b,h)
    {
      GemmP P{};
      P.A = qkv16; P.sAz = bhStride; P.lda = TDN;
      P.BT = qkv16 + (size_t)TB * TH * TS * TDN; P.sBz = bhStride; P.ldb = TDN;
      P.out32 = sc32; P.sCz = (long long)TS * TS; P.ldc = TS;
      P.M = TS; P.N = TS; P.K = TDN; P.scale = scale;
      dim3 g(TS / 128, TS / BM, TB * TH);
      wmma_gemm<EPI_SCORES, 128, false, true><<<g, 256, 0, stream>>>(P);
    }
    // masked softmax (fp32) -> p16
    softmax_kernel<<<dim3(TS, TB * TH), 256, 0, stream>>>(sc32, p16, x);
    // O = P . V, batched; V already stored as [DN][S] = B^T
    {
      GemmP P{};
      P.A = p16; P.sAz = (long long)TS * TS; P.lda = TS;
      P.BT = qkv16 + (size_t)2 * TB * TH * TS * TDN; P.sBz = bhStride; P.ldb = TS;
      P.out16 = o16;
      P.M = TS; P.N = TDN; P.K = TS;
      dim3 g(1, TS / BM, TB * TH);
      wmma_gemm<EPI_PV, 64, false, true><<<g, 256, 0, stream>>>(P);
    }
    // h += O . Wo + bo
    {
      GemmP P{};
      P.A = o16; P.lda = TD;
      P.BT = woT + (size_t)l * TD * TD; P.ldb = TD;
      P.bias = bo + (size_t)l * TD;
      P.resid = hbuf; P.ldr = TD;
      P.out32 = hbuf; P.ldc = TD;
      P.M = TB * TS; P.N = TD; P.K = TD;
      dim3 g(TD / 128, (TB * TS) / BM, 1);
      wmma_gemm<EPI_RESID, 128, false, true><<<g, 256, 0, stream>>>(P);
    }
    // LN2 -> a16
    ln_kernel<<<TB * TS, 256, 0, stream>>>(hbuf, ln2g + (size_t)l * TD,
                                           ln2b + (size_t)l * TD, a16, TD);
    // FFN1 + exact GELU -> ff16
    {
      GemmP P{};
      P.A = a16; P.lda = TD;
      P.BT = fc1T + (size_t)l * TDFF * TD; P.ldb = TD;
      P.bias = fc1b + (size_t)l * TDFF;
      P.out16 = ff16; P.ldc = TDFF;
      P.M = TB * TS; P.N = TDFF; P.K = TD;
      dim3 g(TDFF / 128, (TB * TS) / BM, 1);
      wmma_gemm<EPI_GELU, 128, false, true><<<g, 256, 0, stream>>>(P);
    }
    // h += FFN2(ff16)
    {
      GemmP P{};
      P.A = ff16; P.lda = TDFF;
      P.BT = fc2T + (size_t)l * TD * TDFF; P.ldb = TDFF;
      P.bias = fc2b + (size_t)l * TD;
      P.resid = hbuf; P.ldr = TD;
      P.out32 = hbuf; P.ldc = TD;
      P.M = TB * TS; P.N = TD; P.K = TDFF;
      dim3 g(TD / 128, (TB * TS) / BM, 1);
      wmma_gemm<EPI_RESID, 128, false, true><<<g, 256, 0, stream>>>(P);
    }
  }

  // final LN -> a16, then LM head (fp32 logits, N guarded: 50257 % 128 != 0)
  ln_kernel<<<TB * TS, 256, 0, stream>>>(hbuf, lnfg, lnfb, a16, TD);
  {
    GemmP P{};
    P.A = a16; P.lda = TD;
    P.BT = headT; P.ldb = TD;
    P.out32 = out; P.ldc = TV;
    P.M = TB * TS; P.N = TV; P.K = TD;
    dim3 g(cdiv(TV, 128), (TB * TS) / BM, 1);
    wmma_gemm<EPI_F32, 128, true, false><<<g, 256, 0, stream>>>(P);
  }
}
